// Encoder_26061861552804
// MI455X (gfx1250) — matrix operand order repaired and checked by reading, not compiled, not run on hardware
//
#include <hip/hip_runtime.h>

#define N_NODES 100000
#define N_EDGES 1600000
#define C_IN    256
#define C_HID   128
#define BN_EPS  1e-5f

typedef __bf16 bf16_t;
typedef __attribute__((ext_vector_type(16))) __bf16        v16bf;
typedef __attribute__((ext_vector_type(8)))  float         v8f;
typedef __attribute__((ext_vector_type(4)))  unsigned int  v4u;

union V16Frag { v4u u[2]; v16bf v; };

// ---------------------------------------------------------------------------
// Generic fill / convert
// ---------------------------------------------------------------------------
__global__ void fill_kernel(float* __restrict__ p, float v, size_t n) {
    size_t i = (size_t)blockIdx.x * blockDim.x + threadIdx.x;
    size_t stride = (size_t)gridDim.x * blockDim.x;
    for (; i < n; i += stride) p[i] = v;
}

__global__ void f32_to_bf16_kernel(const float* __restrict__ in, bf16_t* __restrict__ out,
                                   size_t n) {
    size_t i = (size_t)blockIdx.x * blockDim.x + threadIdx.x;
    size_t stride = (size_t)gridDim.x * blockDim.x;
    for (; i < n; i += stride) out[i] = (bf16_t)in[i];
}

// deg[dst] += 1  (deg pre-filled with 1.0 for the self loop)
__global__ void deg_kernel(const int* __restrict__ dst, float* __restrict__ deg, int nE) {
    int e = blockIdx.x * blockDim.x + threadIdx.x;
    if (e < nE) atomicAdd(&deg[dst[e]], 1.0f);
}

__global__ void rsqrt_kernel(float* __restrict__ p, int n) {
    int i = blockIdx.x * blockDim.x + threadIdx.x;
    if (i < n) p[i] = rsqrtf(p[i]);
}

// ---------------------------------------------------------------------------
// WMMA GEMM: C[nrows x 128] = A[nrows x K] @ W[K x 128]  (bf16 MAC, f32 acc)
// A is bf16 row-major. W is staged into LDS pre-swizzled into the exact
// per-lane B-fragment order, so a B fragment is one contiguous 32B LDS read.
// FUSED=false: W = one f32 K x 128 matrix (W1)
// FUSED=true : W = two f32 128 x 64 head matrices [W2 | W3]
// One wave -> 16-row strip x 128 cols (8 accumulators). 8 waves / block.
// ---------------------------------------------------------------------------
template <int K, bool FUSED>
__global__ void gcn_gemm(const bf16_t* __restrict__ A,
                         const float* __restrict__ Wa,
                         const float* __restrict__ Wb,
                         float* __restrict__ C, int nrows) {
    // packed layout: [(K/32) kblk][8 ct][32 lane][8 pairs] -> K*128 bf16 total
    __shared__ bf16_t sW[K * 128];                 // K=256 -> 64KB, K=128 -> 32KB
    for (int pidx = threadIdx.x; pidx < K * 64; pidx += blockDim.x) {
        int j    = pidx & 7;
        int ln   = (pidx >> 3) & 31;
        int ct   = (pidx >> 8) & 7;
        int kblk = pidx >> 11;
        int n = ln & 15, hf = ln >> 4;
        int k = kblk * 32 + hf * 16 + 2 * j;
        int col = ct * 16 + n;
        float w0, w1;
        if (!FUSED) {
            w0 = Wa[(size_t)k * 128 + col];
            w1 = Wa[(size_t)(k + 1) * 128 + col];
        } else if (col < 64) {
            w0 = Wa[(size_t)k * 64 + col];
            w1 = Wa[(size_t)(k + 1) * 64 + col];
        } else {
            int c2 = col - 64;
            w0 = Wb[(size_t)k * 64 + c2];
            w1 = Wb[(size_t)(k + 1) * 64 + c2];
        }
        sW[2 * pidx]     = (bf16_t)w0;
        sW[2 * pidx + 1] = (bf16_t)w1;
    }
    __syncthreads();

    const int lane = threadIdx.x & 31;
    const int strip = blockIdx.x * 8 + (threadIdx.x >> 5);
    const int row0 = strip * 16;
    if (row0 >= nrows) return;                     // wave-uniform exit: EXEC stays all-1

    const int m = lane & 15, half = lane >> 4;
    const bf16_t* Arow = A + (size_t)(row0 + m) * K;

    v8f acc[8];
#pragma unroll
    for (int t = 0; t < 8; ++t) acc[t] = (v8f)0.0f;

#pragma unroll
    for (int k0 = 0; k0 < K; k0 += 32) {
        // A fragment: one contiguous 32B run (k0+half*16+{0..15})
        const v4u* Ap = (const v4u*)(Arow + k0 + half * 16);
        V16Frag af;
        af.u[0] = Ap[0];
        af.u[1] = Ap[1];
        v16bf a = af.v;
#pragma unroll
        for (int ct = 0; ct < 8; ++ct) {
            // B fragment: one contiguous 32B LDS read (pre-swizzled)
            const v16bf* Bp =
                (const v16bf*)&sW[(((k0 >> 5) * 8 + ct) * 32 + lane) * 16];
            acc[ct] = __builtin_amdgcn_wmma_f32_16x16x32_bf16(
                false, a, false, *Bp, (short)0, acc[ct], false, false);
        }
    }
    // store: D VGPR j -> row j + 8*half, col = ct*16 + m
#pragma unroll
    for (int ct = 0; ct < 8; ++ct)
#pragma unroll
        for (int j = 0; j < 8; ++j)
            C[(size_t)(row0 + half * 8 + j) * 128 + ct * 16 + m] = acc[ct][j];
}

// ---------------------------------------------------------------------------
// Edge scatter: agg[dst] += feat[src] * dinv[src]*dinv[dst], F = 128.
// One wave per edge, float4 per lane, 4 f32 atomics per lane.
// ---------------------------------------------------------------------------
__global__ void scatter_kernel(const int* __restrict__ src, const int* __restrict__ dst,
                               const float* __restrict__ dinv,
                               const float* __restrict__ feat,
                               float* __restrict__ agg, int nE) {
    int wave = blockIdx.x * (blockDim.x >> 5) + (threadIdx.x >> 5);
    if (wave >= nE) return;
    int lane = threadIdx.x & 31;
    int s = src[wave], d = dst[wave];
    float w = dinv[s] * dinv[d];
    const float4 v = *(const float4*)(feat + (size_t)s * 128 + lane * 4);
    float* ap = agg + (size_t)d * 128 + lane * 4;
    atomicAdd(ap + 0, v.x * w);
    atomicAdd(ap + 1, v.y * w);
    atomicAdd(ap + 2, v.z * w);
    atomicAdd(ap + 3, v.w * w);
}

// ---------------------------------------------------------------------------
// z = agg + h1*selfw + b1 (in place in agg), accumulate per-channel sum/sumsq
// blockDim = 128 (thread == channel), rows strided over grid
// ---------------------------------------------------------------------------
__global__ void combine1_kernel(float* __restrict__ z, const float* __restrict__ h1,
                                const float* __restrict__ dinv, const float* __restrict__ b1,
                                float* __restrict__ stats, int nrows) {
    int c = threadIdx.x;
    float bc = b1[c];
    float sum = 0.f, sq = 0.f;
    for (int r = blockIdx.x; r < nrows; r += gridDim.x) {
        float sw = dinv[r];
        sw *= sw;
        size_t g = (size_t)r * 128 + c;
        float v = z[g] + h1[g] * sw + bc;
        z[g] = v;
        sum += v;
        sq += v * v;
    }
    atomicAdd(&stats[c], sum);
    atomicAdd(&stats[128 + c], sq);
}

__global__ void bnparam_kernel(float* __restrict__ stats, const float* __restrict__ gamma,
                               const float* __restrict__ beta, float invN) {
    int c = threadIdx.x;
    float mean = stats[c] * invN;
    float var = stats[128 + c] * invN - mean * mean;   // biased var, as torch BN
    float scale = gamma[c] * rsqrtf(var + BN_EPS);
    stats[256 + c] = scale;
    stats[384 + c] = beta[c] - mean * scale;
}

// h = relu(z*scale + shift) -> bf16 activations for the head GEMMs
__global__ void bnrelu_kernel(const float* __restrict__ z, const float* __restrict__ stats,
                              bf16_t* __restrict__ hb, size_t total) {
    size_t g = (size_t)blockIdx.x * blockDim.x + threadIdx.x;
    size_t stride = (size_t)gridDim.x * blockDim.x;
    for (; g < total; g += stride) {
        int c = (int)(g & 127);
        float v = z[g] * stats[256 + c] + stats[384 + c];
        hb[g] = (bf16_t)fmaxf(v, 0.0f);
    }
}

// final: v = agg + t*selfw + b; split fused 128 cols into xm (0..63) / x_ (64..127)
__global__ void combine2_kernel(const float* __restrict__ agg, const float* __restrict__ t,
                                const float* __restrict__ dinv,
                                const float* __restrict__ b2, const float* __restrict__ b3,
                                float* __restrict__ out, int nrows) {
    size_t total = (size_t)nrows * 128;
    size_t g = (size_t)blockIdx.x * blockDim.x + threadIdx.x;
    size_t stride = (size_t)gridDim.x * blockDim.x;
    for (; g < total; g += stride) {
        int r = (int)(g >> 7), c = (int)(g & 127);
        float sw = dinv[r];
        sw *= sw;
        float v = agg[g] + t[g] * sw + ((c < 64) ? b2[c] : b3[c - 64]);
        if (c < 64) out[(size_t)r * 64 + c] = v;
        else        out[(size_t)N_NODES * 64 + (size_t)r * 64 + (c - 64)] = v;
    }
}

// ---------------------------------------------------------------------------
extern "C" void kernel_launch(void* const* d_in, const int* in_sizes, int n_in,
                              void* d_out, int out_size, void* d_ws, size_t ws_size,
                              hipStream_t stream) {
    const float* x     = (const float*)d_in[0];
    const int*   ei    = (const int*)d_in[1];
    const float* W1    = (const float*)d_in[2];
    const float* b1    = (const float*)d_in[3];
    const float* gamma = (const float*)d_in[4];
    const float* beta  = (const float*)d_in[5];
    const float* W2    = (const float*)d_in[6];
    const float* b2    = (const float*)d_in[7];
    const float* W3    = (const float*)d_in[8];
    const float* b3    = (const float*)d_in[9];
    float* out = (float*)d_out;

    const int* srcIdx = ei;
    const int* dstIdx = ei + N_EDGES;

    // workspace carve-up (256B aligned)
    char* ws = (char*)d_ws;
    size_t off = 0;
    auto carve = [&](size_t bytes) -> void* {
        void* p = ws + off;
        off += (bytes + 255) & ~(size_t)255;
        return p;
    };
    float*  dinv  = (float*)carve((size_t)N_NODES * sizeof(float));
    float*  bufA  = (float*)carve((size_t)N_NODES * 128 * sizeof(float)); // h1, then t2|t3
    float*  bufB  = (float*)carve((size_t)N_NODES * 128 * sizeof(float)); // agg1/z, then agg2|agg3
    bf16_t* xbf   = (bf16_t*)carve((size_t)N_NODES * C_IN * sizeof(bf16_t));
    bf16_t* hbf   = (bf16_t*)carve((size_t)N_NODES * 128 * sizeof(bf16_t));
    float*  stats = (float*)carve(512 * sizeof(float));

    const size_t NF = (size_t)N_NODES * 128;
    const unsigned gElem = (unsigned)((NF + 255) / 256);
    const unsigned gGemm = (unsigned)((N_NODES / 16 + 7) / 8);   // 6250 strips / 8 waves
    const unsigned gScat = (unsigned)(N_EDGES / 8);              // 8 edge-waves per block

    // --- norm: deg -> d^-1/2 ; x -> bf16 ---
    fill_kernel<<<512, 256, 0, stream>>>(dinv, 1.0f, (size_t)N_NODES);
    fill_kernel<<<4096, 256, 0, stream>>>(bufB, 0.0f, NF);
    fill_kernel<<<1, 256, 0, stream>>>(stats, 0.0f, (size_t)512);
    deg_kernel<<<N_EDGES / 256, 256, 0, stream>>>(dstIdx, dinv, N_EDGES);
    rsqrt_kernel<<<(N_NODES + 255) / 256, 256, 0, stream>>>(dinv, N_NODES);
    f32_to_bf16_kernel<<<8192, 256, 0, stream>>>(x, xbf, (size_t)N_NODES * C_IN);

    // --- layer 1: GEMM -> scatter -> combine+BN+ReLU ---
    gcn_gemm<C_IN, false><<<gGemm, 256, 0, stream>>>(xbf, W1, nullptr, bufA, N_NODES);
    scatter_kernel<<<gScat, 256, 0, stream>>>(srcIdx, dstIdx, dinv, bufA, bufB, N_EDGES);
    combine1_kernel<<<512, 128, 0, stream>>>(bufB, bufA, dinv, b1, stats, N_NODES);
    bnparam_kernel<<<1, 128, 0, stream>>>(stats, gamma, beta, 1.0f / (float)N_NODES);
    bnrelu_kernel<<<gElem, 256, 0, stream>>>(bufB, stats, hbf, NF);

    // --- heads (fused W2|W3): GEMM -> scatter -> combine ---
    fill_kernel<<<4096, 256, 0, stream>>>(bufB, 0.0f, NF);
    gcn_gemm<C_HID, true><<<gGemm, 256, 0, stream>>>(hbf, W2, W3, bufA, N_NODES);
    scatter_kernel<<<gScat, 256, 0, stream>>>(srcIdx, dstIdx, dinv, bufA, bufB, N_EDGES);
    combine2_kernel<<<gElem, 256, 0, stream>>>(bufB, bufA, dinv, b2, b3, out, N_NODES);

    (void)in_sizes; (void)n_in; (void)out_size; (void)ws_size;
}
